// RoPEAttention_61684320305230
// MI455X (gfx1250) — matrix operand order repaired and checked by reading, not compiled, not run on hardware
//
#include <hip/hip_runtime.h>

#define HID   1024
#define HEADS 16
#define HD    64
#define BATCH 4
#define SEQ   2048
#define MROWS (BATCH * SEQ)

typedef __attribute__((ext_vector_type(8)))  __bf16       v8bf;
typedef __attribute__((ext_vector_type(16))) __bf16       v16bf;
typedef __attribute__((ext_vector_type(8)))  float        v8f;
typedef __attribute__((ext_vector_type(4)))  unsigned int v4u;
typedef __attribute__((ext_vector_type(8)))  int          v8i;
typedef __attribute__((ext_vector_type(4)))  int          v4i;

// ---- fragment loaders -------------------------------------------------------
// A operand (16x32 bf16): element i -> k = half*16 + i
static __device__ __forceinline__ v16bf ld_frag_a(const __bf16* rowp, int half) {
  v8bf lo = *(const v8bf*)(rowp + half * 16);
  v8bf hi = *(const v8bf*)(rowp + half * 16 + 8);
  v16bf r;
#pragma unroll
  for (int i = 0; i < 8; ++i) { r[i] = lo[i]; r[i + 8] = hi[i]; }
  return r;
}

// B operand (32x16 bf16): lanes 0-15 hold K=0..15, lanes 16-31 hold K=16..31
// -> element i -> k = half*16 + i : 16 contiguous bf16 per lane (caller adds half*16).
static __device__ __forceinline__ v16bf ld_frag_b(const __bf16* p) {
  v8bf lo = *(const v8bf*)p;
  v8bf hi = *(const v8bf*)(p + 8);
  v16bf r;
#pragma unroll
  for (int i = 0; i < 8; ++i) { r[i] = lo[i]; r[i + 8] = hi[i]; }
  return r;
}

static __device__ __forceinline__ v8f wmma_bf16(v16bf a, v16bf b, v8f c) {
  return __builtin_amdgcn_wmma_f32_16x16x32_bf16(false, a, false, b, (short)0, c,
                                                 false, false);
}

// ---- fp32 -> bf16 conversion ------------------------------------------------
__global__ void cvt_bf16(const float* __restrict__ src, __bf16* __restrict__ dst, int n) {
  int i = blockIdx.x * blockDim.x + threadIdx.x;
  if (i < n) dst[i] = (__bf16)src[i];
}

// ---- NT GEMM: C[M,N] = A[M,K] * W[N,K]^T  (bf16 in, f32 acc) ---------------
// block = 128 threads (4 waves); block tile 64(M) x 64(N); wave tile 16 x 64.
// Software-pipelined: next k-step's fragments load while current WMMAs execute.
__global__ __launch_bounds__(128) void gemm_nt_bf16(
    const __bf16* __restrict__ A, const __bf16* __restrict__ W,
    __bf16* __restrict__ Cb, float* __restrict__ Cf,
    int M, int N, int K, int write_f32) {
  const int lane = threadIdx.x & 31;
  const int wave = threadIdx.x >> 5;
  const int half = lane >> 4;
  const int lr   = lane & 15;
  const int m0 = blockIdx.y * 64 + wave * 16;
  const int n0 = blockIdx.x * 64;

  const __bf16* arow  = A + (size_t)(m0 + lr) * K;
  const __bf16* brow0 = W + (size_t)(n0 +  0 + lr) * K + half * 16;
  const __bf16* brow1 = W + (size_t)(n0 + 16 + lr) * K + half * 16;
  const __bf16* brow2 = W + (size_t)(n0 + 32 + lr) * K + half * 16;
  const __bf16* brow3 = W + (size_t)(n0 + 48 + lr) * K + half * 16;

  v8f acc[4] = {};
  v16bf a  = ld_frag_a(arow, half);
  v16bf b0 = ld_frag_b(brow0);
  v16bf b1 = ld_frag_b(brow1);
  v16bf b2 = ld_frag_b(brow2);
  v16bf b3 = ld_frag_b(brow3);

  for (int k0 = 0; k0 < K; k0 += 32) {
    int kn = (k0 + 32 < K) ? (k0 + 32) : 0;   // last-iter reload is discarded
    v16bf an  = ld_frag_a(arow + kn, half);
    v16bf bn0 = ld_frag_b(brow0 + kn);
    v16bf bn1 = ld_frag_b(brow1 + kn);
    v16bf bn2 = ld_frag_b(brow2 + kn);
    v16bf bn3 = ld_frag_b(brow3 + kn);
    acc[0] = wmma_bf16(a, b0, acc[0]);
    acc[1] = wmma_bf16(a, b1, acc[1]);
    acc[2] = wmma_bf16(a, b2, acc[2]);
    acc[3] = wmma_bf16(a, b3, acc[3]);
    a = an; b0 = bn0; b1 = bn1; b2 = bn2; b3 = bn3;
  }

  // C layout: row m = m0 + half*8 + v, col = n0 + t*16 + lr
#pragma unroll
  for (int t = 0; t < 4; ++t) {
    int col = n0 + t * 16 + lr;
#pragma unroll
    for (int v = 0; v < 8; ++v) {
      size_t idx = (size_t)(m0 + half * 8 + v) * N + col;
      if (write_f32) Cf[idx] = acc[t][v];
      else           Cb[idx] = (__bf16)acc[t][v];
    }
  }
}

// ---- RoPE in place on Q and K; folds 1/sqrt(64) into Q ---------------------
__global__ void rope_qk(__bf16* __restrict__ Q, __bf16* __restrict__ Kk, int total) {
  int idx = blockIdx.x * blockDim.x + threadIdx.x;
  if (idx >= total) return;
  int j = idx & 31;           // rotation pair index 0..31
  int h = (idx >> 5) & 15;    // head
  int m = idx >> 9;           // row in [0, B*L)
  int t = m & (SEQ - 1);      // position
  float fj  = __expf(-9.210340371976184f * (float)j * (1.0f / 32.0f)); // 10000^(-j/32)
  float ang = (float)t * fj;
  float s, c;
  __sincosf(ang, &s, &c);
  size_t base = (size_t)m * HID + h * HD + j;
  float q1 = (float)Q[base], q2 = (float)Q[base + 32];
  Q[base]      = (__bf16)((q1 * c - q2 * s) * 0.125f);
  Q[base + 32] = (__bf16)((q2 * c + q1 * s) * 0.125f);
  float k1 = (float)Kk[base], k2 = (float)Kk[base + 32];
  Kk[base]      = (__bf16)(k1 * c - k2 * s);
  Kk[base + 32] = (__bf16)(k2 * c + k1 * s);
}

// ---- Flash attention: block = (qtile, head, batch), 4 waves ----------------
// V tiles are staged into LDS by the Tensor Data Mover (tensor_load_to_lds),
// fenced with s_wait_tensorcnt + workgroup barrier.
__global__ __launch_bounds__(128) void attn(
    const __bf16* __restrict__ Q, const __bf16* __restrict__ Kt,
    const __bf16* __restrict__ V, __bf16* __restrict__ ctx) {
  __shared__ __align__(16) __bf16 Vlds[32 * HD];        // 32 keys x 64 dims
  __shared__ __align__(16) __bf16 Plds[4][16 * 32];     // per-wave P tile

  const int lane = threadIdx.x & 31;
  const int wave = threadIdx.x >> 5;
  const int half = lane >> 4;
  const int lr   = lane & 15;
  const int b = blockIdx.z, h = blockIdx.y, qt = blockIdx.x;
  const size_t rowbase = (size_t)b * SEQ;
  const int q0 = qt * 64 + wave * 16;

  // LDS byte offset of Vlds: flat LDS address truncates to addr[31:0]
  const unsigned vlds_off = (unsigned)(size_t)(void*)&Vlds[0];

  // Q fragments for both 32-wide d-steps (rows fixed for whole kernel)
  const __bf16* qp = Q + (rowbase + q0 + lr) * HID + h * HD;
  v16bf qa[2];
#pragma unroll
  for (int s2 = 0; s2 < 2; ++s2) qa[s2] = ld_frag_a(qp + s2 * 32, half);

  float runmax[8], runsum[8];
  v8f o[4] = {};
#pragma unroll
  for (int v = 0; v < 8; ++v) { runmax[v] = -1e30f; runsum[v] = 0.0f; }

  for (int j0 = 0; j0 < SEQ; j0 += 32) {
    // ---- TDM: DMA the 32x64 bf16 V tile into LDS (wave 0 issues) ----
    if (wave == 0) {
      unsigned long long ga =
          (unsigned long long)(size_t)(V + (rowbase + j0) * HID + h * HD);
      v4u g0;
      g0[0] = 1u;                                   // count=1, is_restore=0, gather=0
      g0[1] = vlds_off;                             // lds_addr
      g0[2] = (unsigned)(ga & 0xFFFFFFFFu);         // global_addr[31:0]
      g0[3] = (unsigned)((ga >> 32) & 0x01FFFFFFu)  // global_addr[56:32]
              | (2u << 30);                         // type = 2 (image)
      v8i g1;
      g1[0] = (int)(1u << 16);     // workgroup_mask=0, data_size=1 (2 bytes)
      g1[1] = (int)(64u << 16);    // tensor_dim0 = 64 (low 16 bits)
      g1[2] = (int)(32u << 16);    // tensor_dim0 hi=0 | tensor_dim1 = 32 (low 16)
      g1[3] = (int)(64u << 16);    // tensor_dim1 hi=0 | tile_dim0 = 64
      g1[4] = 32;                  // tile_dim1 = 32, tile_dim2 = 0
      g1[5] = 1024;                // tensor_dim0_stride = 1024 (low 32)
      g1[6] = 0;                   // stride hi | tensor_dim1_stride lo
      g1[7] = 0;
      v4i z4 = {0, 0, 0, 0};
      v8i z8 = {0, 0, 0, 0, 0, 0, 0, 0};
      __builtin_amdgcn_tensor_load_to_lds(g0, g1, z4, z4, z8, 0);
      __builtin_amdgcn_s_wait_tensorcnt(0);
    }
    __syncthreads();

    // ---- S = Q * K^T for 32 keys: 4 fragment loads, then 4 WMMAs ----
    const __bf16* krow0 = Kt + (rowbase + j0 + lr) * HID + h * HD + half * 16;
    const __bf16* krow1 = Kt + (rowbase + j0 + 16 + lr) * HID + h * HD + half * 16;
    v16bf kb00 = ld_frag_b(krow0);        // keys 0-15, d 0-31
    v16bf kb10 = ld_frag_b(krow1);        // keys 16-31, d 0-31
    v16bf kb01 = ld_frag_b(krow0 + 32);   // keys 0-15, d 32-63
    v16bf kb11 = ld_frag_b(krow1 + 32);   // keys 16-31, d 32-63
    v8f s0 = {}, s1 = {};
    s0 = wmma_bf16(qa[0], kb00, s0);
    s1 = wmma_bf16(qa[0], kb10, s1);
    s0 = wmma_bf16(qa[1], kb01, s0);
    s1 = wmma_bf16(qa[1], kb11, s1);

    // ---- online softmax per accumulator row (m = half*8 + v) ----
#pragma unroll
    for (int v = 0; v < 8; ++v) {
      float mx = fmaxf(s0[v], s1[v]);
#pragma unroll
      for (int msk = 1; msk < 16; msk <<= 1) mx = fmaxf(mx, __shfl_xor(mx, msk, 32));
      float nm   = fmaxf(runmax[v], mx);
      float corr = __expf(runmax[v] - nm);
      runmax[v]  = nm;
      float p0 = __expf(s0[v] - nm);
      float p1 = __expf(s1[v] - nm);
      float ts = p0 + p1;
#pragma unroll
      for (int msk = 1; msk < 16; msk <<= 1) ts += __shfl_xor(ts, msk, 32);
      runsum[v] = runsum[v] * corr + ts;
      int row = half * 8 + v;
      Plds[wave][row * 32 + lr]      = (__bf16)p0;
      Plds[wave][row * 32 + 16 + lr] = (__bf16)p1;
#pragma unroll
      for (int t = 0; t < 4; ++t) o[t][v] *= corr;
    }

    // re-read P as an A operand (16 queries x 32 keys)
    v16bf pa = ld_frag_a(&Plds[wave][lr * 32], half);

    // ---- O += P * V (B operand gathered from LDS: k = half*16 + i) ----
#pragma unroll
    for (int t = 0; t < 4; ++t) {
      v16bf vb;
#pragma unroll
      for (int i = 0; i < 16; ++i)
        vb[i] = Vlds[(half * 16 + i) * HD + t * 16 + lr];
      o[t] = wmma_bf16(pa, vb, o[t]);
    }
    __syncthreads();
  }

  // finalize: divide by running sum, store context bf16 [B*L, H*Hd]
#pragma unroll
  for (int v = 0; v < 8; ++v) {
    float inv = 1.0f / runsum[v];
    int row = q0 + half * 8 + v;
#pragma unroll
    for (int t = 0; t < 4; ++t)
      ctx[(rowbase + row) * HID + h * HD + t * 16 + lr] = (__bf16)(o[t][v] * inv);
  }
}

// ---- host side --------------------------------------------------------------
extern "C" void kernel_launch(void* const* d_in, const int* in_sizes, int n_in,
                              void* d_out, int out_size, void* d_ws, size_t ws_size,
                              hipStream_t stream) {
  (void)in_sizes; (void)n_in; (void)out_size; (void)ws_size;
  const float* x  = (const float*)d_in[0];
  const float* Wq = (const float*)d_in[1];
  const float* Wk = (const float*)d_in[2];
  const float* Wv = (const float*)d_in[3];
  const float* Wo = (const float*)d_in[4];
  float* out = (float*)d_out;

  const size_t MB = 1ull << 20;
  char* w = (char*)d_ws;
  __bf16* xb  = (__bf16*)(w);             // 16 MB, reused as ctx after projections
  __bf16* Wqb = (__bf16*)(w + 16 * MB);   // 2 MB each
  __bf16* Wkb = (__bf16*)(w + 18 * MB);
  __bf16* Wvb = (__bf16*)(w + 20 * MB);
  __bf16* Wob = (__bf16*)(w + 22 * MB);
  __bf16* Qb  = (__bf16*)(w + 24 * MB);   // 16 MB each
  __bf16* Kb  = (__bf16*)(w + 40 * MB);
  __bf16* Vb  = (__bf16*)(w + 56 * MB);
  __bf16* ctx = xb;

  const int nX = MROWS * HID;     // 8388608
  const int nW = HID * HID;       // 1048576
  cvt_bf16<<<(nX + 255) / 256, 256, 0, stream>>>(x,  xb,  nX);
  cvt_bf16<<<(nW + 255) / 256, 256, 0, stream>>>(Wq, Wqb, nW);
  cvt_bf16<<<(nW + 255) / 256, 256, 0, stream>>>(Wk, Wkb, nW);
  cvt_bf16<<<(nW + 255) / 256, 256, 0, stream>>>(Wv, Wvb, nW);
  cvt_bf16<<<(nW + 255) / 256, 256, 0, stream>>>(Wo, Wob, nW);

  dim3 gg(HID / 64, MROWS / 64);  // (16, 128)
  gemm_nt_bf16<<<gg, 128, 0, stream>>>(xb, Wqb, Qb, nullptr, MROWS, HID, HID, 0);
  gemm_nt_bf16<<<gg, 128, 0, stream>>>(xb, Wkb, Kb, nullptr, MROWS, HID, HID, 0);
  gemm_nt_bf16<<<gg, 128, 0, stream>>>(xb, Wvb, Vb, nullptr, MROWS, HID, HID, 0);

  const int nR = MROWS * HEADS * 32;  // 4194304
  rope_qk<<<(nR + 255) / 256, 256, 0, stream>>>(Qb, Kb, nR);

  dim3 ga(SEQ / 64, HEADS, BATCH);    // (32, 16, 4)
  attn<<<ga, 128, 0, stream>>>(Qb, Kb, Vb, ctx);

  gemm_nt_bf16<<<gg, 128, 0, stream>>>(ctx, Wob, nullptr, out, MROWS, HID, HID, 1);
}
